// Recurrent_7284264534126
// MI455X (gfx1250) — matrix operand order repaired and checked by reading, not compiled, not run on hardware
//
#include <hip/hip_runtime.h>
#include <cstddef>

// ---------------- types for CDNA5 WMMA ----------------
typedef __bf16 bf16_t;
typedef __attribute__((ext_vector_type(16))) __bf16 v16bf;
typedef __attribute__((ext_vector_type(8)))  float  v8f;

#define Bsz 1024
#define Tn  200
#define LXn 190
#define LYn 50
#define Hn  256
#define XDn 64
#define BLK_M 32          // batch rows per workgroup (2 M-tiles)

// ---------------- prep kernel: weights -> bf16 [N][K] ----------------
__global__ void k_transpose(const float* __restrict__ src, bf16_t* __restrict__ dst,
                            int K, int N, int rowOff, int dstK) {
    int idx = blockIdx.x * blockDim.x + threadIdx.x;
    if (idx >= K * N) return;
    int k = idx / N, n = idx - k * N;
    dst[(size_t)(n + rowOff) * dstK + k] = (bf16_t)src[(size_t)k * N + n];
}

// ---------------- fragment helpers ----------------
__device__ __forceinline__ void load_b(v16bf& b, const bf16_t* p) {   // 32B contiguous (global)
    *((uint4*)&b)     = *(const uint4*)p;
    *((uint4*)&b + 1) = *(const uint4*)(p + 8);
}
__device__ __forceinline__ void load_a(v16bf& a, const bf16_t* p) {   // LDS: k..k+15 contiguous
    *((uint4*)&a)     = *(const uint4*)p;
    *((uint4*)&a + 1) = *(const uint4*)(p + 8);
}

// K-loop with double-buffered B fragment; 2 M-tiles per wave.
template <int KK>
__device__ __forceinline__ void gemm2(const bf16_t* __restrict__ w,
                                      const bf16_t* a0, const bf16_t* a1,
                                      v8f& acc0, v8f& acc1) {
    v16bf b;
    load_b(b, w);
#pragma unroll
    for (int kc = 0; kc < KK; kc += 32) {
        v16bf bn;
        if (kc + 32 < KK) load_b(bn, w + kc + 32);   // prefetch next chunk
        v16bf f0, f1;
        load_a(f0, a0 + kc);
        acc0 = __builtin_amdgcn_wmma_f32_16x16x32_bf16(false, f0, false, b, (short)0, acc0, false, false);
        load_a(f1, a1 + kc);
        acc1 = __builtin_amdgcn_wmma_f32_16x16x32_bf16(false, f1, false, b, (short)0, acc1, false, false);
        if (kc + 32 < KK) b = bn;
    }
}

// ---------------- layer 1: Hbf = tanh(A @ W1^T + b1) ----------------
template <int K1, int N1, bool SPLIT>
__device__ __forceinline__ void run_layer1(const bf16_t* __restrict__ W1,
                                           const float* __restrict__ b1a, const float* __restrict__ b1b,
                                           const bf16_t* Abf, bf16_t* Hbf,
                                           int wave, int ln, int half) {
    constexpr int SA = K1 + 8, SH = N1 + 8;
    constexpr int NT = N1 / 16;
#pragma unroll
    for (int i = 0; i < (NT + 7) / 8; ++i) {
        const int nt = wave + 8 * i;
        if ((NT % 8) && nt >= NT) break;
        const int n = nt * 16 + ln;
        v8f acc0 = (v8f)0.0f, acc1 = (v8f)0.0f;
        gemm2<K1>(W1 + (size_t)n * K1 + half * 16,
                  Abf + (ln) * SA + half * 16,
                  Abf + (16 + ln) * SA + half * 16, acc0, acc1);
        const float bb = (SPLIT && i >= 2) ? b1b[n - 256] : b1a[n];
#pragma unroll
        for (int v = 0; v < 8; ++v) {
            Hbf[(v + half * 8) * SH + n]      = (bf16_t)tanhf(acc0[v] + bb);
            Hbf[(16 + v + half * 8) * SH + n] = (bf16_t)tanhf(acc1[v] + bb);
        }
    }
}

// ---------------- layer 2 with stage-specific epilogue ----------------
// EPI: 0 = Euler (h += f*dt, in place)   1 = gates (u->regs, r->LDS bf16)
//      2 = GRU combine (h,std in LDS)    3 = output (global store)
template <int N1, int N2, int EPI>
__device__ __forceinline__ void run_layer2(const bf16_t* __restrict__ W2,
                                           const float* __restrict__ b2a, const float* __restrict__ b2b,
                                           const bf16_t* Hbf, float* hL, float* sL, bf16_t* rL,
                                           v8f (&ureg)[2][2], float dt_t, float* outp,
                                           int wave, int ln, int half) {
    constexpr int SH = N1 + 8;
    constexpr int NT = N2 / 16;
#pragma unroll
    for (int i = 0; i < (NT + 7) / 8; ++i) {
        const int nt = wave + 8 * i;
        if ((NT % 8) && nt >= NT) break;
        const int n2 = nt * 16 + ln;
        const int koff = (EPI == 1 && i >= 2) ? 256 : 0;   // reset gate reads hidden cols 256..511
        v8f acc0 = (v8f)0.0f, acc1 = (v8f)0.0f;
        gemm2<256>(W2 + (size_t)n2 * 256 + half * 16,
                   Hbf + (ln) * SH + koff + half * 16,
                   Hbf + (16 + ln) * SH + koff + half * 16, acc0, acc1);
        const float bb = (EPI == 1 && i >= 2) ? b2b[n2 - 256] : b2a[n2];
#pragma unroll
        for (int mt = 0; mt < 2; ++mt) {
            v8f& acc = mt ? acc1 : acc0;
#pragma unroll
            for (int v = 0; v < 8; ++v) {
                const int m = mt * 16 + v + half * 8;
                const float val = acc[v] + bb;
                if (EPI == 0) {
                    hL[m * Hn + n2] += val * dt_t;                       // hi_ode in place
                } else if (EPI == 1) {
                    const float s = 1.0f / (1.0f + expf(-val));
                    if (i < 2) ureg[i][mt][v] = s;                        // update gate -> registers
                    else       rL[m * (Hn + 8) + (n2 - 256)] = (bf16_t)s; // reset gate -> LDS
                } else if (EPI == 2) {
                    if (i < 2) {                                          // new_y
                        const float uu = ureg[i][mt][v];
                        hL[m * Hn + n2] = (1.0f - uu) * val + uu * hL[m * Hn + n2];
                    } else {                                              // new_y_std
                        const int j = n2 - 256;
                        const float uu = ureg[i - 2][mt][v];
                        const float sv = sL[m * Hn + j];
                        sL[m * Hn + j] = fabsf((1.0f - uu) * fabsf(val) + uu * sv);
                    }
                } else {                                                  // output net
                    outp[(size_t)m * (LYn * XDn) + n2] = val;
                }
            }
        }
    }
}

// ---------------- A-operand staging into bf16 LDS tile ----------------
// MODE 0: A = h (K=256)    MODE 2: [h | std | x]    MODE 3: [h*r | std*r | x]
template <int MODE>
__device__ __forceinline__ void stageA(bf16_t* Abf, const float* hL, const float* sL,
                                       const bf16_t* rL,
                                       const float* __restrict__ x_data,
                                       const float* __restrict__ x_mask,
                                       int rowBase, int xm, int tid) {
    constexpr int K1 = (MODE == 0) ? 256 : 576;
    constexpr int SA = K1 + 8;
    for (int idx = tid; idx < BLK_M * K1; idx += 256) {
        const int row = idx / K1, col = idx - row * K1;
        float v;
        if (MODE == 0) {
            v = hL[row * Hn + col];
        } else if (col < 256) {
            v = hL[row * Hn + col];
            if (MODE == 3) v *= (float)rL[row * (Hn + 8) + col];
        } else if (col < 512) {
            const int c = col - 256;
            v = sL[row * Hn + c];
            if (MODE == 3) v *= (float)rL[row * (Hn + 8) + c];
        } else {
            const int c = col - 512;
            if (xm >= 0) {
                const size_t xo = ((size_t)(rowBase + row) * LXn + xm) * XDn + c;
                v = x_data[xo] * x_mask[xo];
            } else v = 0.0f;
        }
        Abf[row * SA + col] = (bf16_t)v;
    }
}

// ---------------- persistent whole-sequence kernel ----------------
struct Params {
    const bf16_t *Wf1, *Wf2, *Wg1, *Wg2, *Wn1, *Wn2, *Wo1, *Wo2;
    const float *f_b1, *f_b2, *ug_b1, *ug_b2, *rg_b1, *rg_b2, *ns_b1, *ns_b2, *o_b1, *o_b2;
    const float *x_data, *x_mask, *x_time;
    const int *x_idx, *y_idx;
    float* out;
};

__global__ __launch_bounds__(256, 1) void k_scan(Params p) {
    __shared__ float  hL[BLK_M * Hn];                 // recurrent mean state (also hi_ode / new_y)
    __shared__ float  sL[BLK_M * Hn];                 // recurrent std state
    __shared__ alignas(16) bf16_t rL[BLK_M * (Hn + 8)];     // reset gate (bf16)
    __shared__ alignas(16) bf16_t Abf[BLK_M * (576 + 8)];   // staged A operand
    __shared__ alignas(16) bf16_t Hbf[BLK_M * (512 + 8)];   // layer-1 hidden
    __shared__ float dtL[Tn];
    __shared__ int   xmapL[Tn], omapL[Tn];

    const int tid  = threadIdx.x;
    const int lane = tid & 31;
    const int wave = tid >> 5;
    const int half = lane >> 4;
    const int ln   = lane & 15;
    const int rowBase = blockIdx.x * BLK_M;

    // per-workgroup init: state = 0, dt / time maps
    for (int i = tid; i < BLK_M * Hn; i += 256) { hL[i] = 0.0f; sL[i] = 0.0f; }
    for (int t = tid; t < Tn; t += 256) { xmapL[t] = -1; omapL[t] = -1; }
    __syncthreads();
    for (int s = tid; s < LXn; s += 256) xmapL[p.x_idx[s]] = s;
    for (int q = tid; q < LYn; q += 256) omapL[p.y_idx[q]] = q;
    for (int t = tid; t < Tn; t += 256) {
        float d;
        if (t == 0)      d = p.x_time[Tn - 1] - (p.x_time[Tn - 1] + 0.01f); // scan-carry semantics
        else if (t == 1) d = p.x_time[Tn - 1] - p.x_time[0];
        else             d = p.x_time[t - 2] - p.x_time[t - 1];
        dtL[t] = d;
    }
    __syncthreads();

    v8f ureg[2][2];   // update gate, register-resident between gate and combine phases

    for (int t = 0; t < Tn; ++t) {
        const float dt_t = dtL[t];
        const int   xm   = xmapL[t];
        const int   q    = omapL[t];

        // ---- stage 1: f-net + Euler (h <- h + f(h)*dt) ----
        stageA<0>(Abf, hL, sL, rL, p.x_data, p.x_mask, rowBase, xm, tid);
        __syncthreads();
        run_layer1<256, 256, false>(p.Wf1, p.f_b1, p.f_b1, Abf, Hbf, wave, ln, half);
        __syncthreads();
        run_layer2<256, 256, 0>(p.Wf2, p.f_b2, p.f_b2, Hbf, hL, sL, rL, ureg, dt_t, nullptr,
                                wave, ln, half);
        __syncthreads();

        // ---- stage 2: update + reset gates ----
        stageA<2>(Abf, hL, sL, rL, p.x_data, p.x_mask, rowBase, xm, tid);
        __syncthreads();
        run_layer1<576, 512, true>(p.Wg1, p.ug_b1, p.rg_b1, Abf, Hbf, wave, ln, half);
        __syncthreads();
        run_layer2<512, 512, 1>(p.Wg2, p.ug_b2, p.rg_b2, Hbf, hL, sL, rL, ureg, dt_t, nullptr,
                                wave, ln, half);
        __syncthreads();

        // ---- stage 3: candidate + GRU combine ----
        stageA<3>(Abf, hL, sL, rL, p.x_data, p.x_mask, rowBase, xm, tid);
        __syncthreads();
        run_layer1<576, 256, false>(p.Wn1, p.ns_b1, p.ns_b1, Abf, Hbf, wave, ln, half);
        __syncthreads();
        run_layer2<256, 512, 2>(p.Wn2, p.ns_b2, p.ns_b2, Hbf, hL, sL, rL, ureg, dt_t, nullptr,
                                wave, ln, half);
        __syncthreads();

        // ---- stage 4: output net, only at the 50 query times ----
        if (q >= 0) {
            stageA<0>(Abf, hL, sL, rL, p.x_data, p.x_mask, rowBase, xm, tid);
            __syncthreads();
            run_layer1<256, 256, false>(p.Wo1, p.o_b1, p.o_b1, Abf, Hbf, wave, ln, half);
            __syncthreads();
            float* outp = p.out + ((size_t)rowBase * LYn + q) * XDn;
            run_layer2<256, 64, 3>(p.Wo2, p.o_b2, p.o_b2, Hbf, hL, sL, rL, ureg, dt_t, outp,
                                   wave, ln, half);
            __syncthreads();
        }
    }
}

// ---------------- host-side orchestration ----------------
extern "C" void kernel_launch(void* const* d_in, const int* in_sizes, int n_in,
                              void* d_out, int out_size, void* d_ws, size_t ws_size,
                              hipStream_t stream) {
    const float* x_data = (const float*)d_in[0];
    const float* x_mask = (const float*)d_in[1];
    const float* x_time = (const float*)d_in[2];
    const int*   x_idx  = (const int*)d_in[3];
    const int*   y_idx  = (const int*)d_in[4];
    const float* ug_w1 = (const float*)d_in[5],  *ug_b1 = (const float*)d_in[6];
    const float* ug_w2 = (const float*)d_in[7],  *ug_b2 = (const float*)d_in[8];
    const float* rg_w1 = (const float*)d_in[9],  *rg_b1 = (const float*)d_in[10];
    const float* rg_w2 = (const float*)d_in[11], *rg_b2 = (const float*)d_in[12];
    const float* ns_w1 = (const float*)d_in[13], *ns_b1 = (const float*)d_in[14];
    const float* ns_w2 = (const float*)d_in[15], *ns_b2 = (const float*)d_in[16];
    const float* out_w1 = (const float*)d_in[17], *out_b1 = (const float*)d_in[18];
    const float* out_w2 = (const float*)d_in[19], *out_b2 = (const float*)d_in[20];
    const float* f_w1 = (const float*)d_in[21], *f_b1 = (const float*)d_in[22];
    const float* f_w2 = (const float*)d_in[23], *f_b2 = (const float*)d_in[24];

    char* ws = (char*)d_ws;
    size_t off = 0;
    auto take = [&](size_t bytes) -> char* {
        char* rp = ws + off;
        off = (off + bytes + 255) & ~(size_t)255;
        return rp;
    };
    bf16_t* Wf1 = (bf16_t*)take((size_t)256 * 256 * 2);
    bf16_t* Wf2 = (bf16_t*)take((size_t)256 * 256 * 2);
    bf16_t* Wg1 = (bf16_t*)take((size_t)512 * 576 * 2);  // [ug_w1 ; rg_w1]
    bf16_t* Wg2 = (bf16_t*)take((size_t)512 * 256 * 2);  // [ug_w2 ; rg_w2]
    bf16_t* Wn1 = (bf16_t*)take((size_t)256 * 576 * 2);
    bf16_t* Wn2 = (bf16_t*)take((size_t)512 * 256 * 2);
    bf16_t* Wo1 = (bf16_t*)take((size_t)256 * 256 * 2);
    bf16_t* Wo2 = (bf16_t*)take((size_t)64  * 256 * 2);

    auto g256 = [](int n) { return dim3((n + 255) / 256); };
    k_transpose<<<g256(256 * 256), 256, 0, stream>>>(f_w1, Wf1, 256, 256, 0, 256);
    k_transpose<<<g256(256 * 256), 256, 0, stream>>>(f_w2, Wf2, 256, 256, 0, 256);
    k_transpose<<<g256(576 * 256), 256, 0, stream>>>(ug_w1, Wg1, 576, 256, 0,   576);
    k_transpose<<<g256(576 * 256), 256, 0, stream>>>(rg_w1, Wg1, 576, 256, 256, 576);
    k_transpose<<<g256(256 * 256), 256, 0, stream>>>(ug_w2, Wg2, 256, 256, 0,   256);
    k_transpose<<<g256(256 * 256), 256, 0, stream>>>(rg_w2, Wg2, 256, 256, 256, 256);
    k_transpose<<<g256(576 * 256), 256, 0, stream>>>(ns_w1, Wn1, 576, 256, 0, 576);
    k_transpose<<<g256(256 * 512), 256, 0, stream>>>(ns_w2, Wn2, 256, 512, 0, 256);
    k_transpose<<<g256(256 * 256), 256, 0, stream>>>(out_w1, Wo1, 256, 256, 0, 256);
    k_transpose<<<g256(256 * 64),  256, 0, stream>>>(out_w2, Wo2, 256, 64,  0, 256);

    Params p;
    p.Wf1 = Wf1; p.Wf2 = Wf2; p.Wg1 = Wg1; p.Wg2 = Wg2;
    p.Wn1 = Wn1; p.Wn2 = Wn2; p.Wo1 = Wo1; p.Wo2 = Wo2;
    p.f_b1 = f_b1; p.f_b2 = f_b2;
    p.ug_b1 = ug_b1; p.ug_b2 = ug_b2; p.rg_b1 = rg_b1; p.rg_b2 = rg_b2;
    p.ns_b1 = ns_b1; p.ns_b2 = ns_b2;
    p.o_b1 = out_b1; p.o_b2 = out_b2;
    p.x_data = x_data; p.x_mask = x_mask; p.x_time = x_time;
    p.x_idx = x_idx; p.y_idx = y_idx;
    p.out = (float*)d_out;

    // one persistent kernel: each workgroup owns 32 batch rows for the full 200-step scan
    k_scan<<<dim3(Bsz / BLK_M), dim3(256), 0, stream>>>(p);
}
